// FuturePredictor_33543694581863
// MI455X (gfx1250) — compile-verified, matrix operand order checked by reading, not run on hardware
//
#include <hip/hip_runtime.h>
#include <stdint.h>

#define B_DIM   64
#define C_DIM   256
#define T_DIM   512
#define K_PRED  12
#define N_NEG   15
#define COPIES  16
#define OFF0    16
#define TT      32                 // t-tile per workgroup (2 WMMA M-tiles)
#define NCOL    43                 // TT + K_PRED - 1 target columns

typedef __attribute__((ext_vector_type(16))) __bf16 v16bf;
typedef __attribute__((ext_vector_type(8)))  float  v8f;

union V16 {
    v16bf v;
    unsigned short u[16];
    uint4 q[2];
};

__device__ __forceinline__ unsigned short f2bf_hi(float f) {
    unsigned u = __float_as_uint(f);
    unsigned r = u + 0x7FFFu + ((u >> 16) & 1u);   // round-to-nearest-even
    return (unsigned short)(r >> 16);
}
__device__ __forceinline__ float bfbits2f(unsigned short s) {
    return __uint_as_float(((unsigned)s) << 16);
}
// K index inside a 16x32 bf16 WMMA operand tile for (vgpr v, half h, lane-group g)
__device__ __forceinline__ int kidx(int v, int h, int g) {
    return 2 * v + ((v >= 4) ? 8 : 0) + h + 8 * g;
}

// ---------------------------------------------------------------------------
// Kernel 0: pre-swizzle w[c,o,k] (fp32) into bf16 hi/lo arrays laid out as the
// per-lane WMMA B-operand image: [k][kstep][o_chunk][lane][16 halves].
// ---------------------------------------------------------------------------
__global__ void __launch_bounds__(256) wswz(const float* __restrict__ w,
                                            unsigned short* __restrict__ whi,
                                            unsigned short* __restrict__ wlo) {
    int p = blockIdx.x * 256 + threadIdx.x;      // 0 .. 786431
    int j     = p & 15;
    int lane  = (p >> 4) & 31;
    int chunk = (p >> 9) & 15;
    int kstep = (p >> 13) & 7;
    int k     = p >> 16;                          // 0..11
    int c = kstep * 32 + kidx(j >> 1, j & 1, lane >> 4);
    int o = chunk * 16 + (lane & 15);
    float f = w[(c * C_DIM + o) * K_PRED + k];
    unsigned short hi = f2bf_hi(f);
    whi[p] = hi;
    wlo[p] = f2bf_hi(f - bfbits2f(hi));
}

// ---------------------------------------------------------------------------
// Fused kernel: one workgroup per (b, 32-wide t tile).
//   Phase 1: z_pred via split-bf16 WMMA (hi*hi + hi*lo + lo*hi), 2 M-tiles.
//   Phase 2: Σ_o z_pred[t,o,k] * targets[n,o,t+k+16], targets staged to LDS
//            with per-lane async global->LDS gathers (ASYNCcnt-tracked).
// ---------------------------------------------------------------------------
__global__ void __launch_bounds__(256) fused(const float* __restrict__ z,
                                             const float* __restrict__ negz,
                                             const float* __restrict__ bias,
                                             const int*   __restrict__ idxs,
                                             const unsigned short* __restrict__ whi,
                                             const unsigned short* __restrict__ wlo,
                                             float* __restrict__ out) {
    __shared__ __align__(16) unsigned short a_hi[2][8 * 32 * 16];   // 16 KB
    __shared__ __align__(16) unsigned short a_lo[2][8 * 32 * 16];   // 16 KB
    __shared__ __align__(16) float zp[2][K_PRED][16][16];            // 24 KB  [mt][k][t][o]
    __shared__ __align__(16) float tgt[16 * NCOL * 16];              // 43 KB  [n][col][c]
    __shared__ unsigned long long tgtbase[16 * NCOL];                // 5.5 KB

    const int tid  = threadIdx.x;
    const int b    = blockIdx.y;
    const int t0   = blockIdx.x * TT;
    const int lane = tid & 31;
    const int wave = __builtin_amdgcn_readfirstlane(tid) >> 5;       // scalar wave id

    // --- Stage A operands: z[b, :, t0:t0+32] -> hi/lo bf16, WMMA-A-swizzled ---
    {
        int kstep = tid >> 5;
        int t = lane & 15, g = lane >> 4;
#pragma unroll
        for (int mt = 0; mt < 2; ++mt) {
            const float* zb = z + (size_t)b * C_DIM * T_DIM + t0 + mt * 16 + t;
#pragma unroll
            for (int j = 0; j < 16; ++j) {
                int c = kstep * 32 + kidx(j >> 1, j & 1, g);
                float f = zb[(size_t)c * T_DIM];
                unsigned short hi = f2bf_hi(f);
                a_hi[mt][(kstep * 32 + lane) * 16 + j] = hi;
                a_lo[mt][(kstep * 32 + lane) * 16 + j] = f2bf_hi(f - bfbits2f(hi));
            }
        }
    }

    // --- gather base pointers for target columns; zero-fill永 invalid rows once ---
    for (int e = tid; e < 16 * NCOL; e += 256) {
        int n2 = e / NCOL, col = e - n2 * NCOL;
        int tcol = t0 + OFF0 + col;
        unsigned long long bp = 0;
        if (tcol < T_DIM) {
            if (n2 == 0) {
                bp = (unsigned long long)(uintptr_t)(z + (size_t)b * C_DIM * T_DIM + tcol);
            } else {
                int idx = idxs[b * (N_NEG * T_DIM) + (n2 - 1) * T_DIM + tcol];
                bp = (unsigned long long)(uintptr_t)
                     (negz + (size_t)(idx >> 9) * C_DIM * T_DIM + (idx & (T_DIM - 1)));
            }
        } else {
            // invalid column: staged value is always 0 for every chunk
#pragma unroll
            for (int c = 0; c < 16; ++c) tgt[e * 16 + c] = 0.0f;
        }
        tgtbase[e] = bp;
    }

    const int tloc = tid & 15;
    const int n    = tid >> 4;

    float acc2[2][K_PRED] = {};

    __syncthreads();

    const int nk  = (wave < 4) ? 2 : 1;     // scalar: waves 0-3 own {w, w+8}; 4-7 own {w}
    const int ks0 = wave;
    const int ks1 = wave + 8;

    for (int chunk = 0; chunk < 16; ++chunk) {
        const int c0 = chunk * 16;

        // ---- Phase 1 GEMM for this 16-wide o chunk, 2 M-tiles ----
        v8f acc[2][2] = {};
        for (int kstep = 0; kstep < 8; ++kstep) {
            V16 Ah[2], Al[2];
#pragma unroll
            for (int mt = 0; mt < 2; ++mt) {
                const uint4* ap  = (const uint4*)&a_hi[mt][(kstep * 32 + lane) * 16];
                const uint4* alp = (const uint4*)&a_lo[mt][(kstep * 32 + lane) * 16];
                Ah[mt].q[0] = ap[0];  Ah[mt].q[1] = ap[1];
                Al[mt].q[0] = alp[0]; Al[mt].q[1] = alp[1];
            }
#pragma unroll
            for (int kk = 0; kk < 2; ++kk) {
                if (kk >= nk) break;                 // scalar branch
                int k = kk ? ks1 : ks0;
                size_t boff = (((size_t)(k * 8 + kstep) * 16 + chunk) * 32 + lane) * 16;
                V16 Bh, Bl;
                const uint4* bp  = (const uint4*)&whi[boff];
                const uint4* blp = (const uint4*)&wlo[boff];
                Bh.q[0] = bp[0];  Bh.q[1] = bp[1];
                Bl.q[0] = blp[0]; Bl.q[1] = blp[1];
#pragma unroll
                for (int mt = 0; mt < 2; ++mt) {
                    acc[mt][kk] = __builtin_amdgcn_wmma_f32_16x16x32_bf16(
                        false, Ah[mt].v, false, Bh.v, (short)0, acc[mt][kk], false, false);
                    acc[mt][kk] = __builtin_amdgcn_wmma_f32_16x16x32_bf16(
                        false, Ah[mt].v, false, Bl.v, (short)0, acc[mt][kk], false, false);
                    acc[mt][kk] = __builtin_amdgcn_wmma_f32_16x16x32_bf16(
                        false, Al[mt].v, false, Bh.v, (short)0, acc[mt][kk], false, false);
                }
            }
        }

        __syncthreads();   // previous chunk's phase-2 readers done with zp/tgt

        // ---- stage target slab via async global->LDS gathers (overlaps zp scatter)
        //      tgt[n][col][c] = targets[n, c0+c, t0+16+col]
#pragma unroll 1
        for (int it = 0; it < NCOL; ++it) {          // NCOL*256 == 16*NCOL*16
            int id = it * 256 + tid;
            int rowi = id >> 4, c = id & 15;         // rowi = n*NCOL + col
            unsigned long long bptr = tgtbase[rowi];
            if (bptr) {
                unsigned long long gaddr =
                    bptr + ((unsigned long long)(unsigned)((c0 + c) * T_DIM) << 2);
                unsigned ldsoff = (unsigned)(uintptr_t)&tgt[id];   // LDS byte offset
                asm volatile("global_load_async_to_lds_b32 %0, %1, off"
                             :: "v"(ldsoff), "v"(gaddr)
                             : "memory");
            }
        }

        // ---- scatter D tiles (+bias) to zp: D layout lane%16 = o, reg r -> t = r + 8*(lane/16)
        {
            int oloc = lane & 15;
            float bo = bias[c0 + oloc];
#pragma unroll
            for (int kk = 0; kk < 2; ++kk) {
                if (kk >= nk) break;
                int k = kk ? ks1 : ks0;
#pragma unroll
                for (int mt = 0; mt < 2; ++mt) {
#pragma unroll
                    for (int r = 0; r < 8; ++r) {
                        int t = r + ((lane >> 4) << 3);
                        zp[mt][k][t][oloc] = acc[mt][kk][r] + bo;
                    }
                }
            }
        }

        // all of this wave's async LDS writes must land before the barrier
        asm volatile("s_wait_asynccnt 0x0" ::: "memory");
        __syncthreads();

        // ---- Phase 2: acc2[m][k] += <zp[m][k][tloc][:], tgt[n][m*16+tloc+k][:]> ----
#pragma unroll
        for (int m = 0; m < 2; ++m) {
            int bc = m * 16 + tloc;
#pragma unroll
            for (int k = 0; k < K_PRED; ++k) {
                const float4* zr = (const float4*)&zp[m][k][tloc][0];
                const float4* tr = (const float4*)&tgt[(n * NCOL + bc + k) * 16];
                float s = acc2[m][k];
#pragma unroll
                for (int q = 0; q < 4; ++q) {
                    float4 a4 = zr[q], b4 = tr[q];
                    s = fmaf(a4.x, b4.x, s);
                    s = fmaf(a4.y, b4.y, s);
                    s = fmaf(a4.z, b4.z, s);
                    s = fmaf(a4.w, b4.w, s);
                }
                acc2[m][k] = s;
            }
        }
    }

    // ---- write predictions: step-concatenated 'tbn' layout, scaled by 5 ----
    int kbase[K_PRED];
    {
        int base = 0;
#pragma unroll
        for (int i = 0; i < K_PRED; ++i) {
            kbase[i] = base;
            base += (T_DIM - OFF0 - i) * B_DIM * COPIES;
        }
    }
#pragma unroll
    for (int m = 0; m < 2; ++m) {
        int tg = t0 + m * 16 + tloc;
#pragma unroll
        for (int k = 0; k < K_PRED; ++k) {
            if (tg + k + OFF0 < T_DIM) {
                size_t pos = (size_t)kbase[k] + ((size_t)tg * B_DIM + b) * COPIES + n;
                out[pos] = 5.0f * acc2[m][k];
            }
        }
    }
}

// zero the int32 labels region (0 bit pattern identical for int and float)
__global__ void __launch_bounds__(256) zlabels(int* out, int start, int count) {
    int i = blockIdx.x * 256 + threadIdx.x;
    if (i < count) out[start + i] = 0;
}

extern "C" void kernel_launch(void* const* d_in, const int* in_sizes, int n_in,
                              void* d_out, int out_size, void* d_ws, size_t ws_size,
                              hipStream_t stream) {
    const float* z    = (const float*)d_in[0];
    const float* negz = (const float*)d_in[1];
    const float* w    = (const float*)d_in[2];
    const float* bias = (const float*)d_in[3];
    const int*   idxs = (const int*)d_in[4];

    unsigned short* whi = (unsigned short*)d_ws;       // 786432 halves = 1.5 MB
    unsigned short* wlo = whi + 786432;                // 1.5 MB more

    // 1) pre-swizzle weights into WMMA B-operand hi/lo bf16 images
    wswz<<<786432 / 256, 256, 0, stream>>>(w, whi, wlo);

    // 2) fused predictor GEMM + NCE contraction
    dim3 grid(T_DIM / TT, B_DIM);     // 16 x 64 workgroups
    fused<<<grid, 256, 0, stream>>>(z, negz, bias, idxs, whi, wlo, (float*)d_out);

    // 3) labels = zeros
    const int npred = 6027264;   // sum_i (T-16-i)*B*COPIES
    const int nlab  = 376704;    // npred / COPIES
    zlabels<<<(nlab + 255) / 256, 256, 0, stream>>>((int*)d_out, npred, nlab);
}
